// GRUGenerator_7911329759449
// MI455X (gfx1250) — matrix operand order repaired and checked by reading, not compiled, not run on hardware
//
#include <hip/hip_runtime.h>

// ============================================================================
// Bidirectional 4-layer GRU (B=128, D=128, H=256, T=512) for MI455X / gfx1250.
//
// Strategy:
//   * Precompute xg = x @ Wih^T + (bih + bhh[r,z]) per layer/direction as a
//     large parallel WMMA GEMM (f16 in, f32 acc), output pre-swizzled into
//     WMMA C-fragment layout so the serial recurrence loads it with one
//     b128 per lane per tile.
//   * Recurrence: 4 WGs per direction (8 waves each). Wave w owns M-tile w;
//     WG q owns H-tiles 4q..4q+3 (x3 gates -> 12 accum tiles, gate triples
//     register-aligned). h kept as f32 master (swizzled) + f16 shadow
//     (row-major, WMMA A operand), both in global memory shared by the 4 WGPs.
//     The WG's 96 KB slice of Whh is staged ONCE into dynamic LDS (async
//     global->LDS, pre-swizzled per-fragment) so the 512-step serial chain
//     reads weights from LDS only. Per-step sync: s_cluster_barrier fast
//     path, atomic-counter fallback.
//   * xg GEMM stages A/B tiles via CDNA5 async global->LDS (ASYNCcnt),
//     double buffered.
//
// Workspace requirement: ~345 MB (xg f16 buffers dominate).
// ============================================================================

typedef __attribute__((ext_vector_type(16))) _Float16 v16h;
typedef __attribute__((ext_vector_type(8)))  _Float16 v8h;
typedef __attribute__((ext_vector_type(8)))  float    v8f;

#define TSTEPS 512
#define NB     128   // batch
#define HID    256   // hidden
#define G3     768   // 3*H
#define NL     4

__device__ __forceinline__ v16h cat16(v8h lo, v8h hi) {
  return __builtin_shufflevector(lo, hi, 0,1,2,3,4,5,6,7,8,9,10,11,12,13,14,15);
}
__device__ __forceinline__ float sigm(float x) { return 1.0f / (1.0f + __expf(-x)); }

// ---- CDNA5 async global->LDS copy (tracked by ASYNCcnt) --------------------
__device__ __forceinline__ void async_g2l_b128(unsigned lds_off, const void* gptr) {
  asm volatile("global_load_async_to_lds_b128 %0, %1, off"
               :: "v"(lds_off), "v"(gptr) : "memory");
}
__device__ __forceinline__ void wait_async_le3() {
  asm volatile("s_wait_asynccnt 0x3" ::: "memory");
}
__device__ __forceinline__ void wait_async_0() {
  asm volatile("s_wait_asynccnt 0x0" ::: "memory");
}

// ---- cross-workgroup step barrier ------------------------------------------
// Fast path: CDNA5 cluster barrier (all participating WGs in one cluster).
// Fallback (clusterID==0, i.e. plain HIP dispatch): monotonic atomic counter.
__device__ __forceinline__ void group_barrier(unsigned* cnt, unsigned expected) {
  __threadfence();           // release our global stores (h16/h32/xout)
  __syncthreads();
  if (__builtin_amdgcn_cluster_id_x() != 0) {
    __builtin_amdgcn_s_cluster_barrier();   // s_barrier_signal/wait -3
  } else if (threadIdx.x == 0) {
    __hip_atomic_fetch_add(cnt, 1u, __ATOMIC_RELEASE, __HIP_MEMORY_SCOPE_AGENT);
    while (__hip_atomic_load(cnt, __ATOMIC_ACQUIRE, __HIP_MEMORY_SCOPE_AGENT) < expected)
      __builtin_amdgcn_s_sleep(2);
  }
  __syncthreads();
  __threadfence();           // acquire: invalidate stale L0 lines before reloading h
}

// ============================================================================
// Weight / input conversion helpers
// ============================================================================
__global__ void cvt_f32_f16(const float* __restrict__ s, _Float16* __restrict__ d, int n) {
  int i = blockIdx.x * blockDim.x + threadIdx.x;
  if (i < n) d[i] = (_Float16)s[i];
}

// biasx[l][dir][768] = bih + (bhh for r,z gates only; bhh_n stays in recurrence)
__global__ void build_bias(const float* __restrict__ bih0, const float* __restrict__ bhh0,
                           const float* __restrict__ bihL, const float* __restrict__ bhhL,
                           float* __restrict__ biasx) {
  int i = blockIdx.x * blockDim.x + threadIdx.x;
  if (i >= NL * 2 * G3) return;
  int j = i % G3;
  int d = (i / G3) & 1;
  int l = i / (2 * G3);
  float bi = (l == 0) ? bih0[d * G3 + j] : bihL[((l - 1) * 2 + d) * G3 + j];
  float bh = (l == 0) ? bhh0[d * G3 + j] : bhhL[((l - 1) * 2 + d) * G3 + j];
  biasx[i] = bi + ((j < 2 * HID) ? bh : 0.0f);
}

// ============================================================================
// xg GEMM: C[M x 768] = A[M x K](f16) * W^T + bias, written as f16 in
// WMMA-C-swizzled tiles: xg[((tm*48)+tc)*256 + lane*8 + v].
// WG tile: 128(M) x 64(N); 8 waves as 2(M) x 4(N); chunk K=32, double-buffered
// LDS fed by async global->LDS b128 copies.
// ============================================================================
__global__ __launch_bounds__(256) void xg_gemm(
    const _Float16* __restrict__ A,     // [Mtiles*16][K] row-major
    const _Float16* __restrict__ Wall,  // [2][768][K] (f16)
    const float*    __restrict__ biasAll, // [2][768]
    _Float16*       __restrict__ xgAll, // [2][...] swizzled out
    int K, size_t xgDirStride)
{
  const int dir = blockIdx.z;
  const _Float16* W    = Wall + (size_t)dir * G3 * K;
  const float*    bias = biasAll + (size_t)dir * G3;
  _Float16*       xg   = xgAll + (size_t)dir * xgDirStride;

  __shared__ _Float16 Ab[2][128][40];   // rows padded to 40 halfs (20 dwords)
  __shared__ _Float16 Bb[2][64][40];

  const int tid  = threadIdx.x;
  const int lane = tid & 31;
  const int wave = tid >> 5;
  const int msub = wave & 1;            // 0..1 (64 rows each)
  const int nsub = wave >> 1;           // 0..3 (16 cols each)
  const int l15  = lane & 15;
  const int ahb  = (lane < 16) ? 0 : 16;// A-operand contiguous base
  const int bkb  = (lane < 16) ? 0 : 16;// B-operand contiguous base

  const size_t rowBase = (size_t)blockIdx.x * 128;
  const int    colBase = blockIdx.y * 64;

  v8f acc[4];
#pragma unroll
  for (int i = 0; i < 4; ++i) acc[i] = (v8f)0.0f;

  const int nk = K >> 5;

  auto stage = [&](int buf, int kc) {
    const int kpos = kc * 32;
#pragma unroll
    for (int i = 0; i < 2; ++i) {              // A: 128 rows x 4 x 16B segs
      int e = tid + 256 * i;
      int r = e >> 2, s = e & 3;
      unsigned lo = (unsigned)(size_t)(const void*)&Ab[buf][r][s * 8];
      async_g2l_b128(lo, (const void*)(A + (rowBase + r) * (size_t)K + kpos + s * 8));
    }
    {                                          // B: 64 rows x 4 x 16B segs
      int r = tid >> 2, s = tid & 3;
      unsigned lo = (unsigned)(size_t)(const void*)&Bb[buf][r][s * 8];
      async_g2l_b128(lo, (const void*)(W + (size_t)(colBase + r) * K + kpos + s * 8));
    }
  };

  stage(0, 0);
  for (int kc = 0; kc < nk; ++kc) {
    const int buf = kc & 1;
    if (kc + 1 < nk) { stage(buf ^ 1, kc + 1); wait_async_le3(); }
    else             { wait_async_0(); }
    __syncthreads();

    // B fragment: 32 contiguous halfs of weight row (col) nsub*16+l15
    v16h bf = cat16(*(const v8h*)&Bb[buf][nsub * 16 + l15][bkb],
                    *(const v8h*)&Bb[buf][nsub * 16 + l15][bkb + 8]);
#pragma unroll
    for (int amt = 0; amt < 4; ++amt) {
      const int ar = msub * 64 + amt * 16 + l15;
      v16h af = cat16(*(const v8h*)&Ab[buf][ar][ahb],
                      *(const v8h*)&Ab[buf][ar][ahb + 8]);
      acc[amt] = __builtin_amdgcn_wmma_f32_16x16x32_f16(
          false, af, false, bf, (short)0, acc[amt], false, false);
    }
    __syncthreads();
  }

  // epilogue: + bias, cvt f16, store swizzled tile (one b128 per lane per tile)
  const int   col = colBase + nsub * 16 + l15;
  const float bv  = bias[col];
  const size_t tc = (size_t)blockIdx.y * 4 + nsub;
#pragma unroll
  for (int amt = 0; amt < 4; ++amt) {
    size_t tm = rowBase / 16 + msub * 4 + amt;
    v8h o;
#pragma unroll
    for (int v = 0; v < 8; ++v) o[v] = (_Float16)(acc[amt][v] + bv);
    *(v8h*)&xg[(tm * 48 + tc) * 256 + (size_t)lane * 8] = o;
  }
}

// ============================================================================
// Recurrence: grid (4, 2) = 4 WGPs per direction. Wave w -> M-tile w;
// WG q -> H-tiles 4q..4q+3, so its 3 gate tiles per H-tile share lane layout.
// The WG's 12 Whh column-tiles (96 KB f16) are staged once into dynamic LDS
// in per-fragment order: Bs[((tile*8+kk)*32+lane)*16 .. +15]  (32B/lane,
// lane-linear => conflict-free ds_load per fragment).
// ============================================================================
__global__ __launch_bounds__(256) void gru_rec(
    const _Float16* __restrict__ whh,     // [2][768][256] f16, this layer
    const _Float16* __restrict__ xgAll,   // [2][...] swizzled
    size_t xgDirStride,
    const float*    __restrict__ h0,      // (2L, B, H) f32
    const float*    __restrict__ bhhLay,  // [2][768] f32, this layer
    _Float16*       __restrict__ xout,    // [T][128][512] f16
    _Float16*       __restrict__ h16All,  // [2][128][256] f16 shadow
    float*          __restrict__ h32All,  // [2][8][16][32][8] f32 master
    unsigned*       __restrict__ barAll,  // per-dir counters (64B apart)
    int layer)
{
  extern __shared__ _Float16 Bs[];        // [12][8][32][16] = 98304 B

  const int dir  = blockIdx.y;
  const int q    = blockIdx.x;            // H-tile group 0..3
  const int wave = threadIdx.x >> 5;
  const int lane = threadIdx.x & 31;
  const int m    = wave;                  // M-tile 0..7
  const int l15  = lane & 15;
  const int ahb  = (lane < 16) ? 0 : 16;  // A contiguous base
  const int rofs = (lane >= 16) ? 8 : 0;  // C/D row offset

  const _Float16* W   = whh   + (size_t)dir * G3 * HID;
  const _Float16* xg  = xgAll + (size_t)dir * xgDirStride;
  _Float16*       h16 = h16All + (size_t)dir * NB * HID;
  float*          h32 = h32All + (size_t)dir * (8 * 16 * 32 * 8);
  unsigned*       bar = barAll + dir * 16;
  const float*    bhn = bhhLay + (size_t)dir * G3 + 2 * HID;   // bhh_n

  float bn[4];
#pragma unroll
  for (int c = 0; c < 4; ++c) bn[c] = bhn[(q * 4 + c) * 16 + l15];

  // ---- stage this WG's Whh slice into LDS, pre-swizzled per-fragment ----
  // 12 tiles x 8 kk x 64 16B-segs = 6144 async b128 transfers, 24 per thread.
#pragma unroll 4
  for (int i = 0; i < 24; ++i) {
    const int e    = threadIdx.x + 256 * i;   // 0..6143
    const int s    = e & 63;                  // seg within (tile,kk)
    const int kk   = (e >> 6) & 7;
    const int tile = e >> 9;                  // 0..11 = g*4 + c
    const int g    = tile >> 2, c = tile & 3;
    const int tc   = g * 16 + q * 4 + c;
    const int lp   = s >> 1;                  // producing lane 0..31
    const int half = s & 1;
    const _Float16* src = W + (size_t)(tc * 16 + (lp & 15)) * HID
                            + kk * 32 + ((lp >= 16) ? 16 : 0) + half * 8;
    unsigned dst = (unsigned)(size_t)(const void*)&Bs[(size_t)e * 8];
    async_g2l_b128(dst, (const void*)src);
  }
  wait_async_0();

  // ---- init h from h0 (f32 master swizzled + f16 shadow row-major) ----
  const float* h0d = h0 + (size_t)(2 * layer + dir) * NB * HID;
#pragma unroll
  for (int c = 0; c < 4; ++c) {
    const int hc = q * 4 + c;
#pragma unroll
    for (int v = 0; v < 8; ++v) {
      const int row = m * 16 + v + rofs;
      const int col = hc * 16 + l15;
      const float hv = h0d[(size_t)row * HID + col];
      h32[(((size_t)m * 16 + hc) * 32 + lane) * 8 + v] = hv;
      h16[(size_t)row * HID + col] = (_Float16)hv;
    }
  }
  group_barrier(bar, (unsigned)((layer * TSTEPS + 1) * 4));

  for (int tt = 0; tt < TSTEPS; ++tt) {
    const int t    = dir ? (TSTEPS - 1 - tt) : tt;
    const int teff = (layer == 0) ? 0 : t;

    // A fragments (full K=256) from the f16 shadow of h (global, cross-WGP)
    v16h af[8];
#pragma unroll
    for (int kk = 0; kk < 8; ++kk) {
      const _Float16* ap = h16 + (size_t)(m * 16 + l15) * HID + kk * 32 + ahb;
      af[kk] = cat16(*(const v8h*)ap, *(const v8h*)(ap + 8));
    }

    v8f acc[3][4];
#pragma unroll
    for (int g = 0; g < 3; ++g)
#pragma unroll
      for (int c = 0; c < 4; ++c) acc[g][c] = (v8f)0.0f;

#pragma unroll
    for (int g = 0; g < 3; ++g) {
#pragma unroll
      for (int c = 0; c < 4; ++c) {
        const int tile = g * 4 + c;
#pragma unroll
        for (int kk = 0; kk < 8; ++kk) {
          const _Float16* bp = &Bs[(((size_t)tile * 8 + kk) * 32 + lane) * 16];
          v16h bf = cat16(*(const v8h*)bp, *(const v8h*)(bp + 8));
          acc[g][c] = __builtin_amdgcn_wmma_f32_16x16x32_f16(
              false, af[kk], false, bf, (short)0, acc[g][c], false, false);
        }
      }
    }

    // gate math: r,z,n tiles for H-tile hc share lane layout with acc tiles
#pragma unroll
    for (int c = 0; c < 4; ++c) {
      const int hc = q * 4 + c;
      const size_t tmg = (size_t)teff * 8 + m;
      const v8h xr8 = *(const v8h*)&xg[(tmg * 48 + (0 * 16 + hc)) * 256 + (size_t)lane * 8];
      const v8h xz8 = *(const v8h*)&xg[(tmg * 48 + (1 * 16 + hc)) * 256 + (size_t)lane * 8];
      const v8h xn8 = *(const v8h*)&xg[(tmg * 48 + (2 * 16 + hc)) * 256 + (size_t)lane * 8];
      float* hp = &h32[(((size_t)m * 16 + hc) * 32 + lane) * 8];
      v8f hold = *(const v8f*)hp;
      v8f hnew;
#pragma unroll
      for (int v = 0; v < 8; ++v) {
        const float r = sigm((float)xr8[v] + acc[0][c][v]);
        const float u = sigm((float)xz8[v] + acc[1][c][v]);
        const float n = tanhf((float)xn8[v] + r * (acc[2][c][v] + bn[c]));
        hnew[v] = (1.0f - u) * n + u * hold[v];
      }
      *(v8f*)hp = hnew;

      const int col = hc * 16 + l15;
#pragma unroll
      for (int v = 0; v < 8; ++v) {
        const int row = m * 16 + v + rofs;
        const _Float16 hv = (_Float16)hnew[v];
        h16[(size_t)row * HID + col] = hv;
        xout[((size_t)t * NB + row) * (2 * HID) + dir * HID + col] = hv;
      }
    }

    if (tt < TSTEPS - 1)
      group_barrier(bar, (unsigned)((layer * TSTEPS + tt + 2) * 4));
  }
}

// ============================================================================
// Final FC: out[b][t] = tanh(dot(x[t][b][0:512], fc_w) + fc_b). Wave/output.
// ============================================================================
__global__ __launch_bounds__(256) void fc_out(
    const _Float16* __restrict__ x, const float* __restrict__ fcw,
    const float* __restrict__ fcb, float* __restrict__ out)
{
  const int wave = threadIdx.x >> 5, lane = threadIdx.x & 31;
  const int o = blockIdx.x * 8 + wave;          // 0..65535 = b*T + t
  const int b = o >> 9, t = o & 511;
  const _Float16* xr = x + ((size_t)t * NB + b) * (2 * HID) + lane * 16;
  float s = 0.0f;
#pragma unroll
  for (int j = 0; j < 16; ++j) s += (float)xr[j] * fcw[lane * 16 + j];
#pragma unroll
  for (int off = 16; off; off >>= 1) s += __shfl_xor(s, off);
  if (lane == 0) out[o] = tanhf(s + fcb[0]);
}

// ============================================================================
// Host side
// ============================================================================
extern "C" void kernel_launch(void* const* d_in, const int* in_sizes, int n_in,
                              void* d_out, int out_size, void* d_ws, size_t ws_size,
                              hipStream_t stream) {
  (void)in_sizes; (void)n_in; (void)out_size; (void)ws_size;
  const float* z    = (const float*)d_in[0];
  const float* h0   = (const float*)d_in[1];
  const float* wih0 = (const float*)d_in[2];
  const float* whh0 = (const float*)d_in[3];
  const float* bih0 = (const float*)d_in[4];
  const float* bhh0 = (const float*)d_in[5];
  const float* wihL = (const float*)d_in[6];
  const float* whhL = (const float*)d_in[7];
  const float* bihL = (const float*)d_in[8];
  const float* bhhL = (const float*)d_in[9];
  const float* fcw  = (const float*)d_in[10];
  const float* fcb  = (const float*)d_in[11];

  char* ws = (char*)d_ws;
  size_t off = 0;
  auto alloc = [&](size_t bytes) { size_t p = off; off = (off + bytes + 255) & ~(size_t)255; return p; };
  const size_t oBar  = alloc(256);
  const size_t oWih0 = alloc((size_t)2 * G3 * 128 * 2);
  const size_t oWihL = alloc((size_t)3 * 2 * G3 * 512 * 2);
  const size_t oWhh  = alloc((size_t)NL * 2 * G3 * HID * 2);
  const size_t oZ16  = alloc((size_t)128 * 128 * 2);
  const size_t oBias = alloc((size_t)NL * 2 * G3 * 4);
  const size_t oH16  = alloc((size_t)2 * NB * HID * 2);
  const size_t oH32  = alloc((size_t)2 * 8 * 16 * 32 * 8 * 4);
  const size_t xgStride = (size_t)TSTEPS * 8 * 48 * 256;   // halfs per direction
  const size_t oXg   = alloc(2 * xgStride * 2);
  const size_t oXa   = alloc((size_t)TSTEPS * NB * 512 * 2);
  const size_t oXb   = alloc((size_t)TSTEPS * NB * 512 * 2);

  _Float16* Wih0f = (_Float16*)(ws + oWih0);
  _Float16* WihLf = (_Float16*)(ws + oWihL);
  _Float16* Whhf  = (_Float16*)(ws + oWhh);
  _Float16* Z16   = (_Float16*)(ws + oZ16);
  float*    Bias  = (float*)(ws + oBias);
  _Float16* H16   = (_Float16*)(ws + oH16);
  float*    H32   = (float*)(ws + oH32);
  _Float16* Xg    = (_Float16*)(ws + oXg);
  unsigned* Bar   = (unsigned*)(ws + oBar);

  hipMemsetAsync(ws + oBar, 0, 256, stream);

  cvt_f32_f16<<<(2 * G3 * 128 + 255) / 256, 256, 0, stream>>>(wih0, Wih0f, 2 * G3 * 128);
  cvt_f32_f16<<<(3 * 2 * G3 * 512 + 255) / 256, 256, 0, stream>>>(wihL, WihLf, 3 * 2 * G3 * 512);
  cvt_f32_f16<<<(2 * G3 * HID + 255) / 256, 256, 0, stream>>>(whh0, Whhf, 2 * G3 * HID);
  cvt_f32_f16<<<(3 * 2 * G3 * HID + 255) / 256, 256, 0, stream>>>(whhL, Whhf + (size_t)2 * G3 * HID, 3 * 2 * G3 * HID);
  cvt_f32_f16<<<(128 * 128 + 255) / 256, 256, 0, stream>>>(z, Z16, 128 * 128);
  build_bias<<<(NL * 2 * G3 + 255) / 256, 256, 0, stream>>>(bih0, bhh0, bihL, bhhL, Bias);

  _Float16* xcur  = (_Float16*)(ws + oXa);
  _Float16* xnext = (_Float16*)(ws + oXb);

  for (int l = 0; l < NL; ++l) {
    if (l == 0) {
      // x is time-invariant at layer 0: one 8-tile xg shared by all steps
      xg_gemm<<<dim3(1, 12, 2), 256, 0, stream>>>(Z16, Wih0f, Bias, Xg, 128, xgStride);
    } else {
      xg_gemm<<<dim3(512, 12, 2), 256, 0, stream>>>(
          xcur, WihLf + (size_t)(l - 1) * 2 * G3 * 512,
          Bias + (size_t)l * 2 * G3, Xg, 512, xgStride);
    }
    const float* bhhLayer = (l == 0) ? bhh0 : (bhhL + (size_t)(l - 1) * 2 * G3);
    gru_rec<<<dim3(4, 2), 256, 98304 /* 96 KB dynamic LDS */, stream>>>(
        Whhf + (size_t)l * 2 * G3 * HID, Xg, xgStride, h0, bhhLayer,
        xnext, H16, H32, Bar, l);
    _Float16* tmp = xcur; xcur = xnext; xnext = tmp;
  }

  fc_out<<<(NB * TSTEPS) / 8, 256, 0, stream>>>(xcur, fcw, fcb, (float*)d_out);
}
